// MultiGNN_60748017434773
// MI455X (gfx1250) — matrix operand order repaired and checked by reading, not compiled, not run on hardware
//
#include <hip/hip_runtime.h>
#include <hip/hip_bf16.h>

typedef __attribute__((ext_vector_type(16))) _Float16 v16h;
typedef __attribute__((ext_vector_type(8)))  _Float16 v8h;
typedef __attribute__((ext_vector_type(8)))  float    v8f;

#define CIN   16
#define HIN   30
#define CHID  16
#define COUT  32
#define KPAD  160          // 9 taps * 16 ch = 144, padded to 10 taps (tap 9 == 0)
#define FLATF 1568         // 32*7*7
#define MLPH  512
#define MLPO  256
#define GNNH  256
#define EMBD  128
#define NHEAD 2

// ---------------- monotone float<->uint for atomic segment-max -------------
__device__ __forceinline__ unsigned fenc(float x) {
    unsigned b = __float_as_uint(x);
    return (b & 0x80000000u) ? ~b : (b | 0x80000000u);
}
__device__ __forceinline__ float fdec(unsigned k) {
    return __uint_as_float((k & 0x80000000u) ? (k ^ 0x80000000u) : ~k);
}

// ---------------- weight prep: transpose f32[K][N] -> f16[N][K] ------------
__global__ void wconvT(const float* __restrict__ W, _Float16* __restrict__ Wt,
                       int K, int N) {
    int i = blockIdx.x * 256 + threadIdx.x;
    if (i >= K * N) return;
    int k = i / N, n = i % N;
    Wt[(size_t)n * K + k] = (_Float16)W[i];
}

// conv weights [Cout][Cin][3][3] -> f16 [Cout][KPAD], K reordered tap-major:
// k = tap*16 + cc   (tap 9 zero-pad)
__global__ void wconvConv(const float* __restrict__ W, _Float16* __restrict__ Wt,
                          int Cout) {
    int i = blockIdx.x * 256 + threadIdx.x;
    if (i >= Cout * KPAD) return;
    int n = i / KPAD, k = i % KPAD;
    int tap = k >> 4, cc = k & 15;
    Wt[i] = (_Float16)(tap < 9 ? W[n * 144 + cc * 9 + tap] : 0.f);
}

__global__ void zero_u32(unsigned* __restrict__ p, int n) {
    int i = blockIdx.x * 256 + threadIdx.x;
    if (i < n) p[i] = 0u;
}

// ---------------- fused conv1+pool+conv2+pool (WMMA implicit GEMM) ---------
// one block (8 wave32) per image; channel-last zero-padded LDS images make
// every A-fragment two contiguous 16B ds loads with compile-time tap offsets.
// B-fragments (weights) are hoisted to registers; only the single partial
// M-tile per conv uses guarded stores.
__global__ __launch_bounds__(256)
void conv_fused_kernel(const float* __restrict__ states,
                       const _Float16* __restrict__ wt1,   // [16][KPAD] tap-major
                       const _Float16* __restrict__ wt2,   // [32][KPAD] tap-major
                       const float* __restrict__ b1,
                       const float* __restrict__ b2,
                       _Float16* __restrict__ flat) {
    __shared__ _Float16 smem[43088];                 // 86176 bytes
    _Float16* s_pad = smem;                          // 32*32*16 = 16384 (padded img, ch-last)
    _Float16* s_c1  = smem + 16384;                  // 900*16  = 14400 (conv1 out, ch-last)
    _Float16* s_p1  = smem + 30784;                  // 17*17*16 = 4624 (padded pool1, ch-last)
    _Float16* s_w1  = smem + 35408;                  // 16*160 = 2560
    _Float16* s_w2  = smem + 37968;                  // 32*160 = 5120
    _Float16* s_c2  = smem;                          // alias s_pad: 225*32 = 7200

    const int t = threadIdx.x;
    for (int i = t; i < 16384; i += 256) s_pad[i] = (_Float16)0.f;
    for (int i = t; i < 4624; i += 256) s_p1[i] = (_Float16)0.f;
    __syncthreads();
    const float* in = states + (size_t)blockIdx.x * (CIN * HIN * HIN);
    for (int i = t; i < CIN * HIN * HIN; i += 256) {
        int c = i / 900, r = i % 900, y = r / 30, x = r % 30;
        s_pad[((y + 1) * 32 + (x + 1)) * 16 + c] = (_Float16)in[i];
    }
    for (int i = t; i < CHID * KPAD; i += 256) s_w1[i] = wt1[i];
    for (int i = t; i < COUT * KPAD; i += 256) s_w2[i] = wt2[i];
    __syncthreads();

    const int wave = t >> 5, lane = t & 31;
    const int lm = lane & 15, lh = lane >> 4;
    constexpr int OFF1[9] = {-33, -32, -31, -1, 0, 1, 31, 32, 33};   // dy*32+dx
    constexpr int OFF2[9] = {-18, -17, -16, -1, 0, 1, 16, 17, 18};   // dy*17+dx

    // ---- conv1: M=900 px (57 tiles), N=16, K=160 (10 taps x 16ch) ----
    {
        v16h bw[5];
#pragma unroll
        for (int ks = 0; ks < 5; ++ks)
            bw[ks] = __builtin_shufflevector(*(const v8h*)(s_w1 + lm * KPAD + ks * 32 + lh * 8), *(const v8h*)(s_w1 + lm * KPAD + ks * 32 + lh * 8 + 16), 0, 1, 2, 3, 4, 5, 6, 7, 8, 9, 10, 11, 12, 13, 14, 15);
        const float bias = b1[lm];

        auto conv1_tile = [&](int mt, bool guard) {
            const int m  = mt * 16 + lm;
            const int py = m / 30, px = m % 30;
            const _Float16* ab = s_pad + ((py + 1) * 32 + px + 1) * 16 + lh * 8;
            v8f c = {};
#pragma unroll
            for (int ks = 0; ks < 5; ++ks) {
                v8h a0 = *(const v8h*)(ab + OFF1[2 * ks] * 16);
                v8h a1 = {};
                if (2 * ks + 1 < 9) a1 = *(const v8h*)(ab + OFF1[2 * ks + 1] * 16);
                v16h a = __builtin_shufflevector(a0, a1, 0, 1, 2, 3, 4, 5, 6, 7,
                                                 8, 9, 10, 11, 12, 13, 14, 15);
                c = __builtin_amdgcn_wmma_f32_16x16x32_f16(false, a, false, bw[ks],
                                                           (short)0, c, false, false);
            }
            _Float16* op = s_c1 + (mt * 16 + 8 * lh) * 16 + lm;
#pragma unroll
            for (int r = 0; r < 8; ++r) {
                float v = c[r] + bias;
                v = v > 0.f ? v : 0.f;
                if (!guard || (mt * 16 + 8 * lh + r) < 900) op[r * 16] = (_Float16)v;
            }
        };
        for (int it = 0; it < 7; ++it) conv1_tile(wave + it * 8, false);  // tiles 0..55
        if (wave == 0) conv1_tile(56, true);                              // partial tile
    }
    __syncthreads();
    // ---- pool1: 30x30 -> padded 17x17 (ch-last) ----
    for (int i = t; i < CHID * 225; i += 256) {
        int c = i & 15, p = i >> 4, y = p / 15, x = p % 15;
        const _Float16* q = s_c1 + ((2 * y) * 30 + 2 * x) * 16 + c;
        float a0 = (float)q[0], a1 = (float)q[16], a2 = (float)q[480], a3 = (float)q[496];
        s_p1[((y + 1) * 17 + x + 1) * 16 + c] = (_Float16)fmaxf(fmaxf(a0, a1), fmaxf(a2, a3));
    }
    __syncthreads();
    // ---- conv2: M=225 px (15 tiles), N=32 (both 16-col tiles per wave) ----
    {
        v16h bwA[5], bwB[5];
#pragma unroll
        for (int ks = 0; ks < 5; ++ks) {
            bwA[ks] = __builtin_shufflevector(*(const v8h*)(s_w2 + lm * KPAD + ks * 32 + lh * 8), *(const v8h*)(s_w2 + lm * KPAD + ks * 32 + lh * 8 + 16), 0, 1, 2, 3, 4, 5, 6, 7, 8, 9, 10, 11, 12, 13, 14, 15);
            bwB[ks] = __builtin_shufflevector(*(const v8h*)(s_w2 + (16 + lm) * KPAD + ks * 32 + lh * 8), *(const v8h*)(s_w2 + (16 + lm) * KPAD + ks * 32 + lh * 8 + 16), 0, 1, 2, 3, 4, 5, 6, 7, 8, 9, 10, 11, 12, 13, 14, 15);
        }
        const float biasA = b2[lm], biasB = b2[16 + lm];

        auto conv2_tile = [&](int mt, bool guard) {
            const int m  = mt * 16 + lm;
            const int py = m / 15, px = m % 15;
            const _Float16* ab = s_p1 + ((py + 1) * 17 + px + 1) * 16 + lh * 8;
            v8f c0 = {}, c1 = {};
#pragma unroll
            for (int ks = 0; ks < 5; ++ks) {
                v8h a0 = *(const v8h*)(ab + OFF2[2 * ks] * 16);
                v8h a1 = {};
                if (2 * ks + 1 < 9) a1 = *(const v8h*)(ab + OFF2[2 * ks + 1] * 16);
                v16h a = __builtin_shufflevector(a0, a1, 0, 1, 2, 3, 4, 5, 6, 7,
                                                 8, 9, 10, 11, 12, 13, 14, 15);
                c0 = __builtin_amdgcn_wmma_f32_16x16x32_f16(false, a, false, bwA[ks],
                                                            (short)0, c0, false, false);
                c1 = __builtin_amdgcn_wmma_f32_16x16x32_f16(false, a, false, bwB[ks],
                                                            (short)0, c1, false, false);
            }
            _Float16* op = s_c2 + (mt * 16 + 8 * lh) * 32 + lm;
#pragma unroll
            for (int r = 0; r < 8; ++r) {
                float v0 = c0[r] + biasA;
                float v1 = c1[r] + biasB;
                v0 = v0 > 0.f ? v0 : 0.f;
                v1 = v1 > 0.f ? v1 : 0.f;
                if (!guard || (mt * 16 + 8 * lh + r) < 225) {
                    op[r * 32]      = (_Float16)v0;
                    op[r * 32 + 16] = (_Float16)v1;
                }
            }
        };
        conv2_tile(wave, false);                       // tiles 0..7
        int mt = 8 + wave;
        if (mt < 14)       conv2_tile(mt, false);      // tiles 8..13
        else if (mt == 14) conv2_tile(14, true);       // partial tile (wave 6)
    }
    __syncthreads();
    // ---- pool2: 15x15 -> 7x7, write flat [c*49 + y*7 + x] ----
    _Float16* out = flat + (size_t)blockIdx.x * FLATF;
    for (int i = t; i < COUT * 49; i += 256) {
        int cc = i / 49, r = i % 49, y = r / 7, x = r % 7;
        const _Float16* p = s_c2 + ((2 * y) * 15 + 2 * x) * 32 + cc;
        float a0 = (float)p[0], a1 = (float)p[32], a2 = (float)p[480], a3 = (float)p[512];
        out[i] = (_Float16)fmaxf(fmaxf(a0, a1), fmaxf(a2, a3));
    }
}

// ---------------- generic WMMA GEMM: C = A[M,K] * Bt[N,K]^T (+bias)(+relu) --
__global__ __launch_bounds__(256)
void gemm_wmma(const _Float16* __restrict__ A, const _Float16* __restrict__ Bt,
               const float* __restrict__ bias, float* __restrict__ Cf,
               _Float16* __restrict__ Ch, int M, int K, int N, int relu) {
    const int wave = threadIdx.x >> 5, lane = threadIdx.x & 31;
    const int lm = lane & 15, lh = lane >> 4;
    const int ntg = N >> 6;
    const int w = blockIdx.x * 8 + wave;
    const int mt = w / ntg, ng = w % ntg;
    if (mt * 16 >= M) return;
    const int row = mt * 16 + lm;
    const _Float16* arow = A + (size_t)row * K;
    const int nb = ng * 64;
    v8f acc[4] = {};
    for (int k0 = 0; k0 < K; k0 += 32) {
        __builtin_prefetch(arow + k0 + 256, 0, 1);
        v8h alo = *(const v8h*)(arow + k0 + lh * 16);
        v8h ahi = *(const v8h*)(arow + k0 + lh * 16 + 8);
        v16h a = __builtin_shufflevector(alo, ahi, 0, 1, 2, 3, 4, 5, 6, 7,
                                         8, 9, 10, 11, 12, 13, 14, 15);
        const _Float16* bcol = Bt + (size_t)(nb + lm) * K + k0 + lh * 16;
#pragma unroll
        for (int j = 0; j < 4; ++j) {
            v16h b = *(const v16h*)(bcol + (size_t)(j * 16) * K);
            acc[j] = __builtin_amdgcn_wmma_f32_16x16x32_f16(false, a, false, b,
                                                            (short)0, acc[j],
                                                            false, false);
        }
    }
    if (Ch) {
#pragma unroll
        for (int j = 0; j < 4; ++j) {
            int col = nb + j * 16 + lm;
            float bv = bias ? bias[col] : 0.f;
#pragma unroll
            for (int r = 0; r < 8; ++r) {
                int orow = mt * 16 + r + 8 * lh;
                float v = acc[j][r] + bv;
                if (relu) v = fmaxf(v, 0.f);
                Ch[(size_t)orow * N + col] = (_Float16)v;
            }
        }
    } else {
#pragma unroll
        for (int j = 0; j < 4; ++j) {
            int col = nb + j * 16 + lm;
            float bv = bias ? bias[col] : 0.f;
#pragma unroll
            for (int r = 0; r < 8; ++r) {
                int orow = mt * 16 + r + 8 * lh;
                float v = acc[j][r] + bv;
                if (relu) v = fmaxf(v, 0.f);
                Cf[(size_t)orow * N + col] = v;
            }
        }
    }
}

// ---------------- GAT pieces ----------------------------------------------
__global__ void attn_logits(const float* __restrict__ h, const float* __restrict__ asrc,
                            const float* __restrict__ adst, float* __restrict__ als,
                            float* __restrict__ ald, int Nn, int D) {
    int i = blockIdx.x * 256 + threadIdx.x;
    if (i >= Nn * NHEAD) return;
    int node = i / NHEAD, hd = i % NHEAD;
    const float* hp = h + (size_t)node * NHEAD * D + hd * D;
    float ss = 0.f, sd = 0.f;
    for (int d = 0; d < D; ++d) {
        float v = hp[d];
        ss += v * asrc[hd * D + d];
        sd += v * adst[hd * D + d];
    }
    als[i] = ss;
    ald[i] = sd;
}

__global__ void edge_max(const int* __restrict__ src, const int* __restrict__ dst,
                         const float* __restrict__ als, const float* __restrict__ ald,
                         unsigned* __restrict__ mx, int E) {
    int i = blockIdx.x * 256 + threadIdx.x;
    if (i >= E * NHEAD) return;
    int e = i / NHEAD, hd = i % NHEAD;
    float v = als[src[e] * NHEAD + hd] + ald[dst[e] * NHEAD + hd];
    v = v > 0.f ? v : 0.2f * v;
    atomicMax(&mx[dst[e] * NHEAD + hd], fenc(v));
}

__global__ void edge_sum(const int* __restrict__ src, const int* __restrict__ dst,
                         const float* __restrict__ als, const float* __restrict__ ald,
                         const unsigned* __restrict__ mx, float* __restrict__ ssum, int E) {
    int i = blockIdx.x * 256 + threadIdx.x;
    if (i >= E * NHEAD) return;
    int e = i / NHEAD, hd = i % NHEAD;
    int d = dst[e] * NHEAD + hd;
    float v = als[src[e] * NHEAD + hd] + ald[d];
    v = v > 0.f ? v : 0.2f * v;
    atomicAdd(&ssum[d], __expf(v - fdec(mx[d])));
}

// one wave per edge: scatter alpha * h[src] into acc[dst]
__global__ void edge_msg(const int* __restrict__ src, const int* __restrict__ dst,
                         const float* __restrict__ als, const float* __restrict__ ald,
                         const unsigned* __restrict__ mx, const float* __restrict__ ssum,
                         const float* __restrict__ h, float* __restrict__ acc,
                         int E, int D) {
    int gid = blockIdx.x * 256 + threadIdx.x;
    int e = gid >> 5, lane = gid & 31;
    if (e >= E) return;
    int s = src[e], d = dst[e];
    for (int hd = 0; hd < NHEAD; ++hd) {
        int di = d * NHEAD + hd;
        float v = als[s * NHEAD + hd] + ald[di];
        v = v > 0.f ? v : 0.2f * v;
        float alpha = __expf(v - fdec(mx[di])) / (ssum[di] + 1e-16f);
        const float* hp = h + (size_t)s * NHEAD * D + hd * D;
        float* ap = acc + (size_t)d * NHEAD * D + hd * D;
        for (int dd = lane; dd < D; dd += 32) atomicAdd(&ap[dd], alpha * hp[dd]);
    }
}

__global__ void gat_finalize(const float* __restrict__ acc, const float* __restrict__ b,
                             float* __restrict__ outf, _Float16* __restrict__ outh,
                             int Nn, int D, int relu) {
    int i = blockIdx.x * 256 + threadIdx.x;
    if (i >= Nn * D) return;
    int node = i / D, dd = i % D;
    const float* ap = acc + (size_t)node * NHEAD * D;
    float v = 0.5f * (ap[dd] + ap[D + dd]) + b[dd];
    if (relu) v = fmaxf(v, 0.f);
    if (outf) outf[i] = v;
    if (outh) outh[i] = (_Float16)v;
}

// ---------------- launch ---------------------------------------------------
static inline void launch_gemm(const _Float16* A, const _Float16* Bt, const float* bias,
                               float* Cf, _Float16* Ch, int M, int K, int N, int relu,
                               hipStream_t s) {
    int waves = (M / 16) * (N / 64);
    int blocks = (waves + 7) / 8;
    gemm_wmma<<<blocks, 256, 0, s>>>(A, Bt, bias, Cf, Ch, M, K, N, relu);
}

extern "C" void kernel_launch(void* const* d_in, const int* in_sizes, int n_in,
                              void* d_out, int out_size, void* d_ws, size_t ws_size,
                              hipStream_t stream) {
    const float* states = (const float*)d_in[0];
    const int*   eidx   = (const int*)d_in[1];
    const float* c1w = (const float*)d_in[2];
    const float* c1b = (const float*)d_in[3];
    const float* c2w = (const float*)d_in[4];
    const float* c2b = (const float*)d_in[5];
    const float* m1w = (const float*)d_in[6];
    const float* m1b = (const float*)d_in[7];
    const float* m2w = (const float*)d_in[8];
    const float* m2b = (const float*)d_in[9];
    const float* g1W = (const float*)d_in[10];
    const float* g1as = (const float*)d_in[11];
    const float* g1ad = (const float*)d_in[12];
    const float* g1b  = (const float*)d_in[13];
    const float* g2W = (const float*)d_in[14];
    const float* g2as = (const float*)d_in[15];
    const float* g2ad = (const float*)d_in[16];
    const float* g2b  = (const float*)d_in[17];

    const int Nn = in_sizes[0] / (CIN * HIN * HIN);  // 8000
    const int E  = in_sizes[1] / 2;                  // 256000
    const int* src = eidx;
    const int* dst = eidx + E;

    char* ws = (char*)d_ws;
    size_t off = 0;
    auto alloc = [&](size_t bytes) -> char* {
        char* p = ws + off;
        off += (bytes + 255) & ~(size_t)255;
        return p;
    };
    auto maxsz = [](size_t a, size_t b) { return a > b ? a : b; };

    _Float16* wt1  = (_Float16*)alloc((size_t)CHID * KPAD * 2);
    _Float16* wt2  = (_Float16*)alloc((size_t)COUT * KPAD * 2);
    _Float16* wtm1 = (_Float16*)alloc((size_t)FLATF * MLPH * 2);
    _Float16* wtm2 = (_Float16*)alloc((size_t)MLPH * MLPO * 2);
    _Float16* wtg1 = (_Float16*)alloc((size_t)MLPO * NHEAD * GNNH * 2);
    _Float16* wtg2 = (_Float16*)alloc((size_t)GNNH * NHEAD * EMBD * 2);

    float*    als1 = (float*)alloc((size_t)Nn * NHEAD * 4);
    float*    ald1 = (float*)alloc((size_t)Nn * NHEAD * 4);
    unsigned* mx1  = (unsigned*)alloc((size_t)Nn * NHEAD * 4);
    float*    s1   = (float*)alloc((size_t)Nn * NHEAD * 4);
    float*    als2 = (float*)alloc((size_t)Nn * NHEAD * 4);
    float*    ald2 = (float*)alloc((size_t)Nn * NHEAD * 4);
    unsigned* mx2  = (unsigned*)alloc((size_t)Nn * NHEAD * 4);
    float*    s2   = (float*)alloc((size_t)Nn * NHEAD * 4);

    _Float16* feat2 = (_Float16*)alloc((size_t)Nn * MLPO * 2);
    _Float16* featg = (_Float16*)alloc((size_t)Nn * GNNH * 2);

    // overlaid big region 1: flat(25.1MB) -> h1(16.4MB) -> h2(8.2MB)
    char* buf1 = alloc(maxsz((size_t)Nn * FLATF * 2, (size_t)Nn * NHEAD * GNNH * 4));
    _Float16* flat = (_Float16*)buf1;
    float*    h1   = (float*)buf1;
    float*    h2   = (float*)buf1;
    // overlaid big region 2: feat1(8.2MB) -> acc1(16.4MB) -> acc2(8.2MB)
    char* buf2 = alloc(maxsz((size_t)Nn * MLPH * 2, (size_t)Nn * NHEAD * GNNH * 4));
    _Float16* feat1 = (_Float16*)buf2;
    float*    acc1  = (float*)buf2;
    float*    acc2  = (float*)buf2;

    // ---- weight prep (f16, transposed [N][K]) ----
    wconvConv<<<(CHID * KPAD + 255) / 256, 256, 0, stream>>>(c1w, wt1, CHID);
    wconvConv<<<(COUT * KPAD + 255) / 256, 256, 0, stream>>>(c2w, wt2, COUT);
    wconvT<<<((FLATF * MLPH) + 255) / 256, 256, 0, stream>>>(m1w, wtm1, FLATF, MLPH);
    wconvT<<<((MLPH * MLPO) + 255) / 256, 256, 0, stream>>>(m2w, wtm2, MLPH, MLPO);
    wconvT<<<((MLPO * NHEAD * GNNH) + 255) / 256, 256, 0, stream>>>(g1W, wtg1, MLPO, NHEAD * GNNH);
    wconvT<<<((GNNH * NHEAD * EMBD) + 255) / 256, 256, 0, stream>>>(g2W, wtg2, GNNH, NHEAD * EMBD);

    // ---- fused conv tower -> flat f16 ----
    conv_fused_kernel<<<Nn, 256, 0, stream>>>(states, wt1, wt2, c1b, c2b, flat);

    // ---- MLP ----
    launch_gemm(flat, wtm1, m1b, nullptr, feat1, Nn, FLATF, MLPH, 1, stream);
    launch_gemm(feat1, wtm2, m2b, nullptr, feat2, Nn, MLPH, MLPO, 0, stream);

    // ---- GAT layer 1 ----
    launch_gemm(feat2, wtg1, nullptr, h1, nullptr, Nn, MLPO, NHEAD * GNNH, 0, stream);
    attn_logits<<<(Nn * NHEAD + 255) / 256, 256, 0, stream>>>(h1, g1as, g1ad, als1, ald1, Nn, GNNH);
    zero_u32<<<((Nn * NHEAD * GNNH) + 255) / 256, 256, 0, stream>>>((unsigned*)acc1, Nn * NHEAD * GNNH);
    zero_u32<<<((Nn * NHEAD) + 255) / 256, 256, 0, stream>>>(mx1, Nn * NHEAD);
    zero_u32<<<((Nn * NHEAD) + 255) / 256, 256, 0, stream>>>((unsigned*)s1, Nn * NHEAD);
    edge_max<<<((E * NHEAD) + 255) / 256, 256, 0, stream>>>(src, dst, als1, ald1, mx1, E);
    edge_sum<<<((E * NHEAD) + 255) / 256, 256, 0, stream>>>(src, dst, als1, ald1, mx1, s1, E);
    edge_msg<<<((E * 32) + 255) / 256, 256, 0, stream>>>(src, dst, als1, ald1, mx1, s1, h1, acc1, E, GNNH);
    gat_finalize<<<((Nn * GNNH) + 255) / 256, 256, 0, stream>>>(acc1, g1b, nullptr, featg, Nn, GNNH, 1);

    // ---- GAT layer 2 ----
    launch_gemm(featg, wtg2, nullptr, h2, nullptr, Nn, GNNH, NHEAD * EMBD, 0, stream);
    attn_logits<<<(Nn * NHEAD + 255) / 256, 256, 0, stream>>>(h2, g2as, g2ad, als2, ald2, Nn, EMBD);
    zero_u32<<<((Nn * NHEAD * EMBD) + 255) / 256, 256, 0, stream>>>((unsigned*)acc2, Nn * NHEAD * EMBD);
    zero_u32<<<((Nn * NHEAD) + 255) / 256, 256, 0, stream>>>(mx2, Nn * NHEAD);
    zero_u32<<<((Nn * NHEAD) + 255) / 256, 256, 0, stream>>>((unsigned*)s2, Nn * NHEAD);
    edge_max<<<((E * NHEAD) + 255) / 256, 256, 0, stream>>>(src, dst, als2, ald2, mx2, E);
    edge_sum<<<((E * NHEAD) + 255) / 256, 256, 0, stream>>>(src, dst, als2, ald2, mx2, s2, E);
    edge_msg<<<((E * 32) + 255) / 256, 256, 0, stream>>>(src, dst, als2, ald2, mx2, s2, h2, acc2, E, EMBD);
    gat_finalize<<<((Nn * EMBD) + 255) / 256, 256, 0, stream>>>(acc2, g2b, (float*)d_out, nullptr, Nn, EMBD, 0);
}
